// SpatialOCRModule_7112465842401
// MI455X (gfx1250) — matrix operand order repaired and checked by reading, not compiled, not run on hardware
//
#include <hip/hip_runtime.h>

typedef __bf16 bf16_t;
typedef bf16_t v16bf __attribute__((ext_vector_type(16)));
typedef bf16_t v8bf  __attribute__((ext_vector_type(8)));
typedef float  v8f   __attribute__((ext_vector_type(8)));

static __device__ __forceinline__ bf16_t f2bf(float f) {
    unsigned int u = __builtin_bit_cast(unsigned int, f);
    unsigned int r = (u + 0x7FFFu + ((u >> 16) & 1u)) >> 16;
    unsigned short h = (unsigned short)r;
    return __builtin_bit_cast(bf16_t, h);
}
static __device__ __forceinline__ float bf2f(bf16_t b) {
    unsigned short h = __builtin_bit_cast(unsigned short, b);
    unsigned int u = ((unsigned int)h) << 16;
    return __builtin_bit_cast(float, u);
}
static __device__ __forceinline__ bf16_t bfzero() {
    return __builtin_bit_cast(bf16_t, (unsigned short)0);
}
// Load a 16-half WMMA fragment as two b128 loads (p and p+gap must be 16B aligned).
static __device__ __forceinline__ v16bf ldfrag(const bf16_t* p, int gap) {
    v8bf lo = *reinterpret_cast<const v8bf*>(p);
    v8bf hi = *reinterpret_cast<const v8bf*>(p + gap);
    return __builtin_shufflevector(lo, hi, 0, 1, 2, 3, 4, 5, 6, 7,
                                   8, 9, 10, 11, 12, 13, 14, 15);
}
#define WMMA_BF16(a, b, c) \
    __builtin_amdgcn_wmma_f32_16x16x32_bf16(false, (a), false, (b), (short)0, (c), false, false)

// Async DMA one dword from global to LDS (ASYNCcnt-tracked, CDNA5 §08).
static __device__ __forceinline__ void async_ld_b32(void* lds, const void* gptr) {
    unsigned ldsoff = (unsigned)(unsigned long long)lds;       // low 32b = LDS offset
    unsigned long long ga = (unsigned long long)gptr;
    asm volatile("global_load_async_to_lds_b32 %0, %1, off"
                 :: "v"(ldsoff), "v"(ga) : "memory");
}
static __device__ __forceinline__ void wait_async0() {
    asm volatile("s_wait_asynccnt 0x0" ::: "memory");
}

// ---------------------------------------------------------------------------
// Prep kernels
// ---------------------------------------------------------------------------
__global__ void bnfold_kernel(const float* __restrict__ bn,
                              float* __restrict__ s, float* __restrict__ t, int Cn) {
    int i = blockIdx.x * blockDim.x + threadIdx.x;
    if (i < Cn) {
        float g = bn[i], b = bn[Cn + i], m = bn[2 * Cn + i], v = bn[3 * Cn + i];
        float sc = g * rsqrtf(v + 1e-5f);
        s[i] = sc;
        t[i] = b - m * sc;
    }
}

__global__ void f2bf_kernel(const float* __restrict__ in, bf16_t* __restrict__ out, long n) {
    long i = (long)blockIdx.x * blockDim.x + threadIdx.x;
    if (i < n) out[i] = f2bf(in[i]);
}

// wbot[co][ci][kh][kw] -> out[((kh*3+kw)*CO + co)*CI + ci]  (bf16, K-contiguous A)
__global__ void wbot_rearrange_kernel(const float* __restrict__ w, bf16_t* __restrict__ out,
                                      int CO, int CI) {
    long total = (long)CO * CI * 9;
    long i = (long)blockIdx.x * blockDim.x + threadIdx.x;
    if (i < total) {
        long tmp = i;
        int ci = (int)(tmp % CI); tmp /= CI;
        int co = (int)(tmp % CO);
        int kk = (int)(tmp / CO);          // kh*3+kw
        out[i] = f2bf(w[((long)co * CI + ci) * 9 + kk]);
    }
}

// NCHW f32 -> NHWC bf16 transpose, 64x64 tiles through LDS (coalesced both ways)
__global__ __launch_bounds__(256) void nchw_to_nhwc_bf16(
    const float* __restrict__ in, bf16_t* __restrict__ out, int C, long Npix) {
    __shared__ bf16_t tile[64 * 72];
    const int tid = threadIdx.x;
    const long n0 = (long)blockIdx.x * 64;
    const int  c0 = blockIdx.y * 64;
    const long bo = (long)blockIdx.z * C * Npix;
    for (int i = tid; i < 64 * 64; i += 256) {
        int c = i >> 6, n = i & 63;
        tile[c * 72 + n] = f2bf(in[bo + (long)(c0 + c) * Npix + n0 + n]);
    }
    __syncthreads();
    for (int i = tid; i < 64 * 64; i += 256) {
        int n = i >> 6, c = i & 63;
        out[bo + (n0 + n) * C + c0 + c] = tile[c * 72 + n];
    }
}

// ---------------------------------------------------------------------------
// k / v path: tiny (19 proxy columns).  One block per (batch, proxy column).
// ---------------------------------------------------------------------------
__global__ __launch_bounds__(256) void kv_kernel(
    const float* __restrict__ proxy,
    const float* __restrict__ wk1, const float* __restrict__ wk2, const float* __restrict__ wv,
    const float* __restrict__ bnk1, const float* __restrict__ bnk2, const float* __restrict__ bnv,
    float* __restrict__ kout, float* __restrict__ vout, int C, int CK, int K19) {
    __shared__ float px[512];
    __shared__ float t1[256];
    int b  = blockIdx.x / K19;
    int kk = blockIdx.x % K19;
    int tid = threadIdx.x;
    for (int i = tid; i < C; i += 256)
        px[i] = proxy[((long)b * C + i) * K19 + kk];
    __syncthreads();
    int ck = tid;   // CK == 256 == blockDim
    float a1 = 0.f, av = 0.f;
    for (int c = 0; c < C; ++c) {
        float p = px[c];
        a1 += wk1[(long)ck * C + c] * p;
        av += wv [(long)ck * C + c] * p;
    }
    {
        float sc = bnk1[ck] * rsqrtf(bnk1[3 * CK + ck] + 1e-5f);
        float vv = (a1 - bnk1[2 * CK + ck]) * sc + bnk1[CK + ck];
        t1[ck] = vv > 0.f ? vv : 0.f;
    }
    {
        float sc = bnv[ck] * rsqrtf(bnv[3 * CK + ck] + 1e-5f);
        float vv = (av - bnv[2 * CK + ck]) * sc + bnv[CK + ck];
        vout[((long)b * CK + ck) * K19 + kk] = vv > 0.f ? vv : 0.f;
    }
    __syncthreads();
    float a2 = 0.f;
    for (int c = 0; c < CK; ++c)
        a2 += wk2[(long)ck * CK + c] * t1[c];
    {
        float sc = bnk2[ck] * rsqrtf(bnk2[3 * CK + ck] + 1e-5f);
        float vv = (a2 - bnk2[2 * CK + ck]) * sc + bnk2[CK + ck];
        kout[((long)b * CK + ck) * K19 + kk] = vv > 0.f ? vv : 0.f;
    }
}

// ---------------------------------------------------------------------------
// LDS-free WMMA GEMM over NHWC activations, 2x2 register blocking per wave:
//   Out[n, m] = relu(s[m] * sum_k A[m,k] * Bm[n,k] + t[m])      (bf16 out)
// Block = 256 threads = 8 waves (2 x 4); wave tile 32x32 -> block tile 64x128.
// 4 WMMAs per (2 A + 2 B) fragment loads.
// ---------------------------------------------------------------------------
__global__ __launch_bounds__(256) void gemm_bf16_nhwc(
    const bf16_t* __restrict__ A,    // M x K (row-major, K contiguous)
    const bf16_t* __restrict__ Bm,   // Npix x K
    bf16_t* __restrict__ Out,        // Npix x M
    const float* __restrict__ s, const float* __restrict__ t,
    int M, int K, long Npix) {
    const int tid  = threadIdx.x;
    const int lane = tid & 31;
    const int wave = tid >> 5;
    const int wm = wave >> 2, wn = wave & 3;
    const int m0 = blockIdx.x * 64 + wm * 32;            // two M subtiles: m0, m0+16
    const long n0 = (long)blockIdx.y * 128 + wn * 32;    // two N subtiles: n0, n0+16
    const int r = lane & 15, hi = lane >> 4;
    const long nb = (long)blockIdx.z * Npix + n0 + r;

    const bf16_t* Ap0 = A + (long)(m0 + r) * K + hi * 8;
    const bf16_t* Ap1 = Ap0 + (long)16 * K;
    const bf16_t* Bp0 = Bm + nb * K + hi * 8;
    const bf16_t* Bp1 = Bp0 + (long)16 * K;

    v8f acc00 = {0.f, 0.f, 0.f, 0.f, 0.f, 0.f, 0.f, 0.f};
    v8f acc01 = acc00, acc10 = acc00, acc11 = acc00;

    #pragma unroll 2
    for (int kk = 0; kk < K; kk += 32) {
        v16bf a0 = ldfrag(Ap0, 16);
        v16bf a1 = ldfrag(Ap1, 16);
        v16bf b0 = ldfrag(Bp0, 16);
        v16bf b1 = ldfrag(Bp1, 16);
        acc00 = WMMA_BF16(a0, b0, acc00);
        acc01 = WMMA_BF16(a0, b1, acc01);
        acc10 = WMMA_BF16(a1, b0, acc10);
        acc11 = WMMA_BF16(a1, b1, acc11);
        Ap0 += 32; Ap1 += 32; Bp0 += 32; Bp1 += 32;
    }

    #pragma unroll
    for (int mi = 0; mi < 2; ++mi) {
        v8f aA = mi ? acc10 : acc00;
        v8f aB = mi ? acc11 : acc01;
        v8bf oA, oB;
        #pragma unroll
        for (int i = 0; i < 8; ++i) {
            int m = m0 + mi * 16 + hi * 8 + i;
            float sv = s[m], tv = t[m];
            float vA = aA[i] * sv + tv;  vA = vA > 0.f ? vA : 0.f;
            float vB = aB[i] * sv + tv;  vB = vB > 0.f ? vB : 0.f;
            oA[i] = f2bf(vA);
            oB[i] = f2bf(vB);
        }
        bf16_t* OpA = Out + (nb) * M + m0 + mi * 16 + hi * 8;
        bf16_t* OpB = Out + (nb + 16) * M + m0 + mi * 16 + hi * 8;
        *reinterpret_cast<v8bf*>(OpA) = oA;
        *reinterpret_cast<v8bf*>(OpB) = oB;
    }
}

// ---------------------------------------------------------------------------
// Attention: sim = (q^T k)/sqrt(CK), softmax over 19, ctx = v @ sim^T
// k/v staged into LDS with GLOBAL_LOAD_ASYNC_TO_LDS_B32 (ASYNCcnt).
// ---------------------------------------------------------------------------
__global__ __launch_bounds__(256) void attn_kernel(
    const bf16_t* __restrict__ q, const float* __restrict__ kbuf,
    const float* __restrict__ vbuf, bf16_t* __restrict__ ctx,
    int CK, long Npix) {
    constexpr int K19 = 19;
    __shared__ float ks[256 * K19];
    __shared__ float vs[256 * K19];
    const int b = blockIdx.y;
    const long n = (long)blockIdx.x * 256 + threadIdx.x;
    const float* kb = kbuf + (long)b * CK * K19;
    const float* vb = vbuf + (long)b * CK * K19;
    for (int i = threadIdx.x; i < CK * K19; i += 256) {
        async_ld_b32(&ks[i], kb + i);
        async_ld_b32(&vs[i], vb + i);
    }
    wait_async0();
    __syncthreads();
    const bf16_t* qb = q + ((long)b * Npix + n) * CK;
    float sim[K19];
    #pragma unroll
    for (int j = 0; j < K19; ++j) sim[j] = 0.f;
    for (int ck = 0; ck < CK; ck += 8) {
        v8bf chunk = *reinterpret_cast<const v8bf*>(qb + ck);
        #pragma unroll
        for (int u = 0; u < 8; ++u) {
            float qv = bf2f(chunk[u]);
            #pragma unroll
            for (int j = 0; j < K19; ++j) sim[j] += qv * ks[(ck + u) * K19 + j];
        }
    }
    const float scale = rsqrtf((float)CK);
    float mx = -3.4e38f;
    #pragma unroll
    for (int j = 0; j < K19; ++j) { sim[j] *= scale; mx = sim[j] > mx ? sim[j] : mx; }
    float sum = 0.f;
    #pragma unroll
    for (int j = 0; j < K19; ++j) { sim[j] = __expf(sim[j] - mx); sum += sim[j]; }
    float inv = 1.f / sum;
    #pragma unroll
    for (int j = 0; j < K19; ++j) sim[j] *= inv;
    bf16_t* cb = ctx + ((long)b * Npix + n) * CK;
    for (int ck = 0; ck < CK; ck += 8) {
        v8bf o;
        #pragma unroll
        for (int u = 0; u < 8; ++u) {
            float a = 0.f;
            #pragma unroll
            for (int j = 0; j < K19; ++j) a += vs[(ck + u) * K19 + j] * sim[j];
            o[u] = f2bf(a);
        }
        *reinterpret_cast<v8bf*>(cb + ck) = o;
    }
}

// ---------------------------------------------------------------------------
// 3x3 conv (SAME) over concat(ctxout, x) in NHWC, LDS-free implicit GEMM,
// 2x2 register blocking.  Block tile = 64 co x 128 pixels, 9216 WMMAs/block.
// Border lanes substitute a zero B fragment (base offset clamped in-bounds,
// lanes reconverge before the WMMA so EXEC is all ones).
// Output written NCHW fp32 (harness layout).
// ---------------------------------------------------------------------------
__global__ __launch_bounds__(256) void conv3x3_nhwc(
    const bf16_t* __restrict__ Wr,   // [(kh*3+kw)*CO + co]*CI + ci
    const bf16_t* __restrict__ Cx,   // (B, H, W, 512)
    const bf16_t* __restrict__ Xb,   // (B, H, W, 512)
    const float* __restrict__ s, const float* __restrict__ t,
    float* __restrict__ Out, int H, int W, int CO, int CI) {
    const int tid  = threadIdx.x;
    const int lane = tid & 31;
    const int wave = tid >> 5;
    const int wm = wave >> 2, wn = wave & 3;
    const int m0 = blockIdx.x * 64 + wm * 32;          // two M subtiles
    const int wtiles = W >> 7;                         // 128-pixel tiles per row
    const int h  = blockIdx.y / wtiles;
    const int w0 = (blockIdx.y % wtiles) * 128;
    const int r = lane & 15, hi = lane >> 4;
    const int wA = w0 + wn * 32 + r;                   // N subtile 0 pixel column
    const int wB = wA + 16;                            // N subtile 1 pixel column
    const long Np = (long)H * W;
    const long bIn  = (long)blockIdx.z * Np * 512;
    const long bOut = (long)blockIdx.z * CO * Np;

    v16bf bz;
    #pragma unroll
    for (int i = 0; i < 16; ++i) bz[i] = bfzero();

    v8f acc00 = {0.f, 0.f, 0.f, 0.f, 0.f, 0.f, 0.f, 0.f};
    v8f acc01 = acc00, acc10 = acc00, acc11 = acc00;

    for (int kh = 0; kh < 3; ++kh) {
        const int hh = h + kh - 1;
        const bool vh = (hh >= 0) && (hh < H);
        if (vh && hh + 1 < H)   // L2 prefetch of next row's pixels for this lane
            __builtin_prefetch(Cx + bIn + ((long)(hh + 1) * W + wA) * 512, 0, 1);
        for (int kw = 0; kw < 3; ++kw) {
            const int wwA = wA + kw - 1;
            const int wwB = wB + kw - 1;
            const bool okA = vh && (wwA >= 0) && (wwA < W);
            const bool okB = vh && (wwB >= 0) && (wwB < W);
            const long poffA = okA ? bIn + ((long)hh * W + wwA) * 512 : 0;
            const long poffB = okB ? bIn + ((long)hh * W + wwB) * 512 : 0;
            const bf16_t* Ap0 = Wr + ((long)((kh * 3 + kw) * CO + m0 + r)) * CI + hi * 8;
            const bf16_t* Ap1 = Ap0 + (long)16 * CI;
            const bf16_t* BcA = Cx + poffA + hi * 8;
            const bf16_t* BcB = Cx + poffB + hi * 8;
            const bf16_t* BxA = Xb + poffA + hi * 8;
            const bf16_t* BxB = Xb + poffB + hi * 8;
            #pragma unroll 2
            for (int cb = 0; cb < 512; cb += 32) {      // channels 0..511 (ctxout)
                v16bf a0 = ldfrag(Ap0 + cb, 16);
                v16bf a1 = ldfrag(Ap1 + cb, 16);
                v16bf b0 = okA ? ldfrag(BcA + cb, 16) : bz;
                v16bf b1 = okB ? ldfrag(BcB + cb, 16) : bz;
                acc00 = WMMA_BF16(a0, b0, acc00);
                acc01 = WMMA_BF16(a0, b1, acc01);
                acc10 = WMMA_BF16(a1, b0, acc10);
                acc11 = WMMA_BF16(a1, b1, acc11);
            }
            #pragma unroll 2
            for (int cb = 0; cb < 512; cb += 32) {      // channels 512..1023 (x)
                v16bf a0 = ldfrag(Ap0 + 512 + cb, 16);
                v16bf a1 = ldfrag(Ap1 + 512 + cb, 16);
                v16bf b0 = okA ? ldfrag(BxA + cb, 16) : bz;
                v16bf b1 = okB ? ldfrag(BxB + cb, 16) : bz;
                acc00 = WMMA_BF16(a0, b0, acc00);
                acc01 = WMMA_BF16(a0, b1, acc01);
                acc10 = WMMA_BF16(a1, b0, acc10);
                acc11 = WMMA_BF16(a1, b1, acc11);
            }
        }
    }

    #pragma unroll
    for (int mi = 0; mi < 2; ++mi) {
        v8f aA = mi ? acc10 : acc00;
        v8f aB = mi ? acc11 : acc01;
        #pragma unroll
        for (int i = 0; i < 8; ++i) {
            int co = m0 + mi * 16 + hi * 8 + i;
            float sv = s[co], tv = t[co];
            float vA = aA[i] * sv + tv;  vA = vA > 0.f ? vA : 0.f;
            float vB = aB[i] * sv + tv;  vB = vB > 0.f ? vB : 0.f;
            float* row = Out + bOut + (long)co * Np + (long)h * W;
            row[wA] = vA;
            row[wB] = vB;
        }
    }
}

// ---------------------------------------------------------------------------
extern "C" void kernel_launch(void* const* d_in, const int* in_sizes, int n_in,
                              void* d_out, int out_size, void* d_ws, size_t ws_size,
                              hipStream_t stream) {
    (void)in_sizes; (void)n_in; (void)out_size; (void)ws_size;
    const int B = 2, C = 512, CK = 256, CO = 512, CI = 1024, H = 128, W = 256, K19 = 19;
    const long N = (long)H * W;                 // 32768 pixels / batch

    const float* x     = (const float*)d_in[0];
    const float* proxy = (const float*)d_in[1];
    const float* wq1   = (const float*)d_in[2];
    const float* bnq1  = (const float*)d_in[3];
    const float* wq2   = (const float*)d_in[4];
    const float* bnq2  = (const float*)d_in[5];
    const float* wk1   = (const float*)d_in[6];
    const float* bnk1  = (const float*)d_in[7];
    const float* wk2   = (const float*)d_in[8];
    const float* bnk2  = (const float*)d_in[9];
    const float* wv    = (const float*)d_in[10];
    const float* bnv   = (const float*)d_in[11];
    const float* wout  = (const float*)d_in[12];
    const float* bnout = (const float*)d_in[13];
    const float* wbot  = (const float*)d_in[14];
    const float* bnbot = (const float*)d_in[15];
    float* out = (float*)d_out;

    char* ws = (char*)d_ws;
    size_t off = 0;
    auto alloc = [&](size_t bytes) -> char* {
        char* p = ws + off;
        off = (off + bytes + 255) & ~(size_t)255;
        return p;
    };
    bf16_t* xbf    = (bf16_t*)alloc((size_t)B * N * C * 2);      // NHWC
    bf16_t* wq1b   = (bf16_t*)alloc((size_t)CK * C * 2);
    bf16_t* wq2b   = (bf16_t*)alloc((size_t)CK * CK * 2);
    bf16_t* woutb  = (bf16_t*)alloc((size_t)C * CK * 2);
    bf16_t* wbotr  = (bf16_t*)alloc((size_t)9 * CO * CI * 2);
    float*  sq1    = (float*)alloc(CK * 4);
    float*  tq1    = (float*)alloc(CK * 4);
    float*  sq2    = (float*)alloc(CK * 4);
    float*  tq2    = (float*)alloc(CK * 4);
    float*  so     = (float*)alloc(C * 4);
    float*  to     = (float*)alloc(C * 4);
    float*  sb     = (float*)alloc(CO * 4);
    float*  tb     = (float*)alloc(CO * 4);
    bf16_t* q1buf  = (bf16_t*)alloc((size_t)B * N * CK * 2);     // NHWC, reused as ctx
    bf16_t* q2buf  = (bf16_t*)alloc((size_t)B * N * CK * 2);     // NHWC
    float*  kbuf   = (float*)alloc((size_t)B * CK * K19 * 4);
    float*  vbuf   = (float*)alloc((size_t)B * CK * K19 * 4);
    bf16_t* ctxout = (bf16_t*)alloc((size_t)B * N * C * 2);      // NHWC

    // ---- prep ----
    bnfold_kernel<<<dim3(1), 256, 0, stream>>>(bnq1, sq1, tq1, CK);
    bnfold_kernel<<<dim3(1), 256, 0, stream>>>(bnq2, sq2, tq2, CK);
    bnfold_kernel<<<dim3(2), 256, 0, stream>>>(bnout, so, to, C);
    bnfold_kernel<<<dim3(2), 256, 0, stream>>>(bnbot, sb, tb, CO);

    {   long n = (long)CK * C;
        f2bf_kernel<<<dim3((unsigned)((n + 255) / 256)), 256, 0, stream>>>(wq1, wq1b, n); }
    {   long n = (long)CK * CK;
        f2bf_kernel<<<dim3((unsigned)((n + 255) / 256)), 256, 0, stream>>>(wq2, wq2b, n); }
    {   long n = (long)C * CK;
        f2bf_kernel<<<dim3((unsigned)((n + 255) / 256)), 256, 0, stream>>>(wout, woutb, n); }
    {   long n = (long)CO * CI * 9;
        wbot_rearrange_kernel<<<dim3((unsigned)((n + 255) / 256)), 256, 0, stream>>>(wbot, wbotr, CO, CI); }

    // x: NCHW f32 -> NHWC bf16
    nchw_to_nhwc_bf16<<<dim3((unsigned)(N / 64), C / 64, B), 256, 0, stream>>>(x, xbf, C, N);

    // ---- k / v path ----
    kv_kernel<<<dim3(B * K19), 256, 0, stream>>>(proxy, wk1, wk2, wv, bnk1, bnk2, bnv,
                                                 kbuf, vbuf, C, CK, K19);

    // ---- q = cba(cba(x, wq1), wq2)  (WMMA GEMMs, NHWC) ----
    gemm_bf16_nhwc<<<dim3(CK / 64, (unsigned)(N / 128), B), 256, 0, stream>>>(
        wq1b, xbf, q1buf, sq1, tq1, CK, C, N);
    gemm_bf16_nhwc<<<dim3(CK / 64, (unsigned)(N / 128), B), 256, 0, stream>>>(
        wq2b, q1buf, q2buf, sq2, tq2, CK, CK, N);

    // ---- attention (q1buf reused as ctx) ----
    attn_kernel<<<dim3((unsigned)(N / 256), B), 256, 0, stream>>>(
        q2buf, kbuf, vbuf, q1buf, CK, N);

    // ---- ctxout = cba(ctx, wout) ----
    gemm_bf16_nhwc<<<dim3(C / 64, (unsigned)(N / 128), B), 256, 0, stream>>>(
        woutb, q1buf, ctxout, so, to, C, CK, N);

    // ---- bottleneck: 3x3 conv over concat(ctxout, x) + BN + ReLU -> NCHW f32 ----
    conv3x3_nhwc<<<dim3(CO / 64, (unsigned)(H * (W / 128)), B), 256, 0, stream>>>(
        wbotr, ctxout, xbf, sb, tb, out, H, W, CO, CI);
}
